// SimpleGNN_45028437131723
// MI455X (gfx1250) — matrix operand order repaired and checked by reading, not compiled, not run on hardware
//
#include <hip/hip_runtime.h>
#include <cstddef>

#define NND   100000
#define NEDG  1600000
#define FIN   500
#define HID   128
#define NCLS  40
#define K1PAD 512
#define N4PAD 48

typedef __attribute__((ext_vector_type(16))) _Float16 v16h;
typedef __attribute__((ext_vector_type(8)))  _Float16 v8h;
typedef __attribute__((ext_vector_type(8)))  float    v8f;

// ---------------- small utility kernels ----------------

__global__ void k_init_deg(float* __restrict__ deg) {
    int i = blockIdx.x * blockDim.x + threadIdx.x;
    if (i < NND) deg[i] = 1.0f;   // self-loop
}

__device__ __forceinline__ void atomicAddF(float* p, float v) {
    unsafeAtomicAdd(p, v);        // hw global_atomic_add_f32 (L2-resident agg)
}

__global__ void k_deg_acc(const int* __restrict__ dst, float* __restrict__ deg) {
    int e = blockIdx.x * blockDim.x + threadIdx.x;
    if (e < NEDG) atomicAddF(&deg[dst[e]], 1.0f);
}

__global__ void k_rsqrt_inplace(float* __restrict__ deg) {
    int i = blockIdx.x * blockDim.x + threadIdx.x;
    if (i < NND) deg[i] = __frsqrt_rn(deg[i]);   // deg -> dinv
}

__global__ void k_zero_f32(float* __restrict__ p, long long n) {
    long long i = (long long)blockIdx.x * blockDim.x + threadIdx.x;
    if (i < n) p[i] = 0.0f;
}

// W row-major [K x N] f32  ->  WT [NPAD x KPAD] f16, WT[n*KPAD+k] = W[k*N+n] (zero pad)
__global__ void k_conv_wT(const float* __restrict__ W, _Float16* __restrict__ WT,
                          int K, int N, int KPAD, int NPAD) {
    int idx = blockIdx.x * blockDim.x + threadIdx.x;
    if (idx >= KPAD * NPAD) return;
    int n = idx / KPAD, k = idx % KPAD;
    float v = (k < K && n < N) ? W[(size_t)k * N + n] : 0.0f;
    WT[idx] = (_Float16)v;
}

// ---------------- WMMA fragment loaders ----------------
// A 16x32 f16 (MxK): lane(m=lane&15, half=lane>>4) holds k = (i>>3)*16 + half*8 + (i&7)
// B 32x16 f16 (KxN): lane(n=lane&15, half=lane>>4) holds k = (i>>3)*16 + half*8 + (i&7)   (from WT[n][k])
// C/D 16x16 f32: lane n = lane&15; vgpr i -> row = (lane>>4)*8 + i

__device__ __forceinline__ v16h load_bT(const _Float16* __restrict__ WT, int ldk,
                                        int n0, int k0, int lane) {
    int n = lane & 15, half = lane >> 4;
    const _Float16* p = WT + (size_t)(n0 + n) * ldk + k0 + half * 8;
    v8h lo = *(const v8h*)(p);
    v8h hi = *(const v8h*)(p + 16);
    v16h b;
#pragma unroll
    for (int i = 0; i < 8; ++i) { b[i] = lo[i]; b[8 + i] = hi[i]; }
    return b;
}

__device__ __forceinline__ v16h load_a16(const _Float16* __restrict__ rowp,
                                         int k0, int half) {
    v8h lo = *(const v8h*)(rowp + k0 + half * 8);
    v8h hi = *(const v8h*)(rowp + k0 + 16 + half * 8);
    v16h a;
#pragma unroll
    for (int i = 0; i < 8; ++i) { a[i] = lo[i]; a[8 + i] = hi[i]; }
    return a;
}

// ---------------- GEMM layer 1:  [NND x 500]f32 @ [500 x 128] -> f32 ----------------
__global__ void k_gemm_x(const float* __restrict__ X, const _Float16* __restrict__ W1T,
                         float* __restrict__ H) {
    const int lane = threadIdx.x & 31;
    const int wid  = threadIdx.x >> 5;
    const int row0 = (blockIdx.x * (blockDim.x >> 5) + wid) * 16;
    if (row0 >= NND) return;
    const int m = lane & 15, half = lane >> 4;
    const float* rowp = X + (size_t)(row0 + m) * FIN;

    v8f z = {0.f, 0.f, 0.f, 0.f, 0.f, 0.f, 0.f, 0.f};
    v8f acc[8];
#pragma unroll
    for (int t = 0; t < 8; ++t) acc[t] = z;

    for (int k0 = 0; k0 < K1PAD; k0 += 32) {
        v16h a;
        if (k0 < 480) {  // fully in-bounds: vectorized f32 loads, convert to f16
            const float4* p0 = (const float4*)(rowp + k0 + half * 8);
            const float4* p1 = (const float4*)(rowp + k0 + 16 + half * 8);
            float4 f0 = p0[0], f1 = p0[1], f2 = p1[0], f3 = p1[1];
            a[0]=(_Float16)f0.x; a[1]=(_Float16)f0.y; a[2]=(_Float16)f0.z; a[3]=(_Float16)f0.w;
            a[4]=(_Float16)f1.x; a[5]=(_Float16)f1.y; a[6]=(_Float16)f1.z; a[7]=(_Float16)f1.w;
            a[8]=(_Float16)f2.x; a[9]=(_Float16)f2.y; a[10]=(_Float16)f2.z; a[11]=(_Float16)f2.w;
            a[12]=(_Float16)f3.x; a[13]=(_Float16)f3.y; a[14]=(_Float16)f3.z; a[15]=(_Float16)f3.w;
        } else {         // masked tail covering k=480..511 (K=500)
#pragma unroll
            for (int i = 0; i < 16; ++i) {
                int k = k0 + ((i >> 3) << 4) + (half << 3) + (i & 7);
                a[i] = (k < FIN) ? (_Float16)rowp[k] : (_Float16)0.f;
            }
        }
        // batch all B-fragment loads, then run the WMMA chain back-to-back
        v16h bf[8];
#pragma unroll
        for (int t = 0; t < 8; ++t) bf[t] = load_bT(W1T, K1PAD, t * 16, k0, lane);
#pragma unroll
        for (int t = 0; t < 8; ++t)
            acc[t] = __builtin_amdgcn_wmma_f32_16x16x32_f16(
                false, a, false, bf[t], (short)0, acc[t], false, false);
    }
    float* out = H + (size_t)row0 * HID;
#pragma unroll
    for (int t = 0; t < 8; ++t)
#pragma unroll
        for (int i = 0; i < 8; ++i)
            out[(size_t)(half * 8 + i) * HID + t * 16 + m] = acc[t][i];
}

// ---- GEMM layers 2-4: [NND x 128]f16 @ [128 x 16*NT] -> f32 (stride LDO), W in LDS ----
template <int NT, int LDO>
__global__ void k_gemm_h(const _Float16* __restrict__ A, const _Float16* __restrict__ WT,
                         float* __restrict__ H) {
    __shared__ __align__(16) _Float16 sW[NT * 16 * HID];   // 32 KB (NT=8) / 12 KB (NT=3)
    const int lane = threadIdx.x & 31;
    const int wid  = threadIdx.x >> 5;

    // cooperative weight staging (ALL threads participate, before any early-out)
    constexpr int nElem = NT * 16 * HID;
    for (int i = threadIdx.x * 8; i < nElem; i += blockDim.x * 8)
        *(v8h*)(sW + i) = *(const v8h*)(WT + i);
    __syncthreads();

    const int row0 = (blockIdx.x * (blockDim.x >> 5) + wid) * 16;
    if (row0 >= NND) return;
    const int m = lane & 15, half = lane >> 4;
    const _Float16* rowp = A + (size_t)(row0 + m) * HID;

    v8f z = {0.f, 0.f, 0.f, 0.f, 0.f, 0.f, 0.f, 0.f};
    v8f acc[NT];
#pragma unroll
    for (int t = 0; t < NT; ++t) acc[t] = z;

#pragma unroll
    for (int k0 = 0; k0 < HID; k0 += 32) {
        v16h a = load_a16(rowp, k0, half);
        v16h bf[NT];
#pragma unroll
        for (int t = 0; t < NT; ++t) bf[t] = load_bT(sW, HID, t * 16, k0, lane);
#pragma unroll
        for (int t = 0; t < NT; ++t)
            acc[t] = __builtin_amdgcn_wmma_f32_16x16x32_f16(
                false, a, false, bf[t], (short)0, acc[t], false, false);
    }
    float* out = H + (size_t)row0 * LDO;
#pragma unroll
    for (int t = 0; t < NT; ++t)
#pragma unroll
        for (int i = 0; i < 8; ++i)
            out[(size_t)(half * 8 + i) * LDO + t * 16 + m] = acc[t][i];
}

// ---------------- edge scatter: agg[dst] += h[src] * dinv[src]*dinv[dst] ----------------
__global__ void k_scatter128(const int* __restrict__ src, const int* __restrict__ dstv,
                             const float* __restrict__ dinv, const float* __restrict__ h,
                             float* __restrict__ agg) {
    int lane = threadIdx.x & 31;
    long long e = (long long)blockIdx.x * (blockDim.x >> 5) + (threadIdx.x >> 5);
    if (e >= NEDG) return;
    int s = src[e], d = dstv[e];
    float norm = dinv[s] * dinv[d];
    float4 v = ((const float4*)(h + (size_t)s * HID))[lane];   // 32 lanes x 16B = 512B row
    float* a = agg + (size_t)d * HID + lane * 4;
    atomicAddF(a + 0, v.x * norm);
    atomicAddF(a + 1, v.y * norm);
    atomicAddF(a + 2, v.z * norm);
    atomicAddF(a + 3, v.w * norm);
}

__global__ void k_scatter40(const int* __restrict__ src, const int* __restrict__ dstv,
                            const float* __restrict__ dinv, const float* __restrict__ h,
                            float* __restrict__ agg) {
    int lane = threadIdx.x & 31;
    long long e = (long long)blockIdx.x * (blockDim.x >> 5) + (threadIdx.x >> 5);
    if (e >= NEDG) return;
    int s = src[e], d = dstv[e];
    float norm = dinv[s] * dinv[d];
    const float* hs = h + (size_t)s * N4PAD;
    float* a = agg + (size_t)d * N4PAD;
    atomicAddF(a + lane, hs[lane] * norm);
    if (lane < NCLS - 32) atomicAddF(a + 32 + lane, hs[32 + lane] * norm);
}

// ---------------- combine + ReLU + f32->f16 (layers 1-3) ----------------
__global__ void k_combine_relu(const float* __restrict__ h, const float* __restrict__ agg,
                               const float* __restrict__ dinv, const float* __restrict__ b,
                               _Float16* __restrict__ act) {
    long long idx = (long long)blockIdx.x * blockDim.x + threadIdx.x;
    if (idx >= (long long)NND * HID) return;
    int n = (int)(idx >> 7), c = (int)(idx & 127);
    float di = dinv[n];
    float v = agg[idx] + h[idx] * (di * di) + b[c];
    act[idx] = (_Float16)fmaxf(v, 0.0f);
}

// ---------------- final combine + softmax over 40 classes ----------------
__global__ void k_final_softmax(const float* __restrict__ h, const float* __restrict__ agg,
                                const float* __restrict__ dinv, const float* __restrict__ b4,
                                float* __restrict__ out) {
    int n = blockIdx.x * blockDim.x + threadIdx.x;
    if (n >= NND) return;
    float di = dinv[n];
    float sn = di * di;
    const float* hp = h + (size_t)n * N4PAD;
    const float* ap = agg + (size_t)n * N4PAD;
    float l[NCLS];
    float vmax = -3.0e38f;
#pragma unroll
    for (int c = 0; c < NCLS; ++c) {
        float v = ap[c] + hp[c] * sn + b4[c];
        l[c] = v;
        vmax = fmaxf(vmax, v);
    }
    float s = 0.0f;
#pragma unroll
    for (int c = 0; c < NCLS; ++c) {
        float e = __expf(l[c] - vmax);
        l[c] = e;
        s += e;
    }
    float inv = 1.0f / s;
    float* op = out + (size_t)n * NCLS;
#pragma unroll
    for (int c = 0; c < NCLS; ++c) op[c] = l[c] * inv;
}

// ---------------- host launch ----------------
extern "C" void kernel_launch(void* const* d_in, const int* in_sizes, int n_in,
                              void* d_out, int out_size, void* d_ws, size_t ws_size,
                              hipStream_t stream) {
    const float* x  = (const float*)d_in[0];
    const int*   ei = (const int*)d_in[1];
    const int* src = ei;
    const int* dst = ei + NEDG;
    const float* W1 = (const float*)d_in[2]; const float* b1 = (const float*)d_in[3];
    const float* W2 = (const float*)d_in[4]; const float* b2 = (const float*)d_in[5];
    const float* W3 = (const float*)d_in[6]; const float* b3 = (const float*)d_in[7];
    const float* W4 = (const float*)d_in[8]; const float* b4 = (const float*)d_in[9];
    float* out = (float*)d_out;

    char* ws = (char*)d_ws;
    auto alloc = [&](size_t bytes) {
        char* p = ws;
        ws += (bytes + 255) & ~(size_t)255;
        return p;
    };
    float*     dinv = (float*)alloc((size_t)NND * 4);                 // deg -> dinv in-place
    float*     hbuf = (float*)alloc((size_t)NND * HID * 4);
    float*     agg  = (float*)alloc((size_t)NND * HID * 4);
    _Float16*  act  = (_Float16*)alloc((size_t)NND * HID * 2);
    _Float16*  w1T  = (_Float16*)alloc((size_t)HID * K1PAD * 2);
    _Float16*  w2T  = (_Float16*)alloc((size_t)HID * HID * 2);
    _Float16*  w3T  = (_Float16*)alloc((size_t)HID * HID * 2);
    _Float16*  w4T  = (_Float16*)alloc((size_t)N4PAD * HID * 2);

    const int B = 256;
    const int gN    = (NND + B - 1) / B;
    const int gE    = (NEDG + B - 1) / B;
    const int gNH   = (int)(((long long)NND * HID + B - 1) / B);
    const int gRow  = (NND / 16 + 7) / 8;       // 8 waves/block, 16 rows/wave
    const int gEdgeW = (NEDG + 7) / 8;          // 8 waves/block, 1 edge/wave

    // degrees / normalization
    k_init_deg<<<gN, B, 0, stream>>>(dinv);
    k_deg_acc<<<gE, B, 0, stream>>>(dst, dinv);
    k_rsqrt_inplace<<<gN, B, 0, stream>>>(dinv);

    // weight conversion + transpose (tiny)
    k_conv_wT<<<(HID * K1PAD + B - 1) / B, B, 0, stream>>>(W1, w1T, FIN, HID, K1PAD, HID);
    k_conv_wT<<<(HID * HID + B - 1) / B, B, 0, stream>>>(W2, w2T, HID, HID, HID, HID);
    k_conv_wT<<<(HID * HID + B - 1) / B, B, 0, stream>>>(W3, w3T, HID, HID, HID, HID);
    k_conv_wT<<<(HID * N4PAD + B - 1) / B, B, 0, stream>>>(W4, w4T, HID, NCLS, HID, N4PAD);

    // layer 1
    k_gemm_x<<<gRow, B, 0, stream>>>(x, w1T, hbuf);
    k_zero_f32<<<gNH, B, 0, stream>>>(agg, (long long)NND * HID);
    k_scatter128<<<gEdgeW, B, 0, stream>>>(src, dst, dinv, hbuf, agg);
    k_combine_relu<<<gNH, B, 0, stream>>>(hbuf, agg, dinv, b1, act);

    // layer 2
    k_gemm_h<8, HID><<<gRow, B, 0, stream>>>(act, w2T, hbuf);
    k_zero_f32<<<gNH, B, 0, stream>>>(agg, (long long)NND * HID);
    k_scatter128<<<gEdgeW, B, 0, stream>>>(src, dst, dinv, hbuf, agg);
    k_combine_relu<<<gNH, B, 0, stream>>>(hbuf, agg, dinv, b2, act);

    // layer 3
    k_gemm_h<8, HID><<<gRow, B, 0, stream>>>(act, w3T, hbuf);
    k_zero_f32<<<gNH, B, 0, stream>>>(agg, (long long)NND * HID);
    k_scatter128<<<gEdgeW, B, 0, stream>>>(src, dst, dinv, hbuf, agg);
    k_combine_relu<<<gNH, B, 0, stream>>>(hbuf, agg, dinv, b3, act);

    // layer 4 (out width 40, padded to 48) + softmax
    k_gemm_h<3, N4PAD><<<gRow, B, 0, stream>>>(act, w4T, hbuf);
    {
        long long n4 = (long long)NND * N4PAD;
        k_zero_f32<<<(int)((n4 + B - 1) / B), B, 0, stream>>>(agg, n4);
    }
    k_scatter40<<<gEdgeW, B, 0, stream>>>(src, dst, dinv, hbuf, agg);
    k_final_softmax<<<gN, B, 0, stream>>>(hbuf, agg, dinv, b4, out);
}
